// OfficialCoreHTGNN_56495999811605
// MI455X (gfx1250) — matrix operand order repaired and checked by reading, not compiled, not run on hardware
//
#include <hip/hip_runtime.h>
#include <hip/hip_bf16.h>
#include <math.h>

// ---------------- problem constants (match reference) ----------------
#define T_STEPS 4
#define N_DEV   25000
#define N_SEN   50000
#define F_IN    64
#define H_DIM   64
#define GH_DIM  128
#define E_DS    600000
#define E_SD    600000
#define E_DD    400000

typedef __attribute__((ext_vector_type(16))) _Float16 v16h;
typedef __attribute__((ext_vector_type(8)))  float    v8f;
typedef unsigned int u32;
typedef __attribute__((ext_vector_type(4))) u32 v4u;
typedef __attribute__((ext_vector_type(8))) int v8i_t;
typedef __attribute__((ext_vector_type(4))) int v4i_t;

// Tensor Data Mover availability (device pass only; host pass parses fallback)
#if defined(__AMDGCN__) && __has_builtin(__builtin_amdgcn_tensor_load_to_lds) && \
    __has_builtin(__builtin_amdgcn_s_wait_tensorcnt)
#define USE_TDM 1
#else
#define USE_TDM 0
#endif

__device__ __forceinline__ float fast_rcp(float x) {
#if defined(__AMDGCN__) && __has_builtin(__builtin_amdgcn_rcpf)
  return __builtin_amdgcn_rcpf(x);
#else
  return 1.0f / x;
#endif
}

// =====================================================================
// WMMA fragment loaders (wave32; A and B both use contiguous k = 16*(lane/16) + e)
// =====================================================================

// A-matrix 16x32 f16 tile from f32 row-major A[M,64], K window [k0,k0+32).
// lane L: m = L%16 ; K base = 16*(L/16); elems 0..15 -> K=base+e.
// Optional per-row scale = 1/max(cnt[m],1) (fused SAGE mean), rows >= M load 0.
__device__ __forceinline__ v16h load_a_frag(const float* __restrict__ A, int M,
                                            int m0, int lane, int k0,
                                            const float* __restrict__ rowcnt) {
  int m       = m0 + (lane & 15);
  int valid   = (m < M);
  int mm      = valid ? m : (M - 1);
  float scale = valid ? 1.0f : 0.0f;
  if (rowcnt) scale *= fast_rcp(fmaxf(rowcnt[mm], 1.0f));
  const float* row = A + (size_t)mm * H_DIM + k0;
  int base = (lane >> 4) * 16;
  v16h a;
#pragma unroll
  for (int e = 0; e < 8; ++e) a[e]     = (_Float16)(row[base + e] * scale);
#pragma unroll
  for (int e = 0; e < 8; ++e) a[e + 8] = (_Float16)(row[8 + base + e] * scale);
  return a;
}

// B-matrix 32x16 f16 tile from global W (B[k,n] = W[n,k], W row-major [64,64]).
__device__ __forceinline__ v16h load_b_frag(const float* __restrict__ W,
                                            int n0, int lane, int k0) {
  int n = n0 + (lane & 15);
  const float* row = W + (size_t)n * H_DIM + k0 + ((lane >> 4) * 16);
  v16h b;
#pragma unroll
  for (int e = 0; e < 16; ++e) b[e] = (_Float16)row[e];
  return b;
}

// B-matrix 32x16 f16 tile from an LDS-staged 16x64 row slice (this wave's rows).
__device__ __forceinline__ v16h load_b_frag_lds(const float* tile16x64,
                                                int lane, int k0) {
  const float* row = tile16x64 + (size_t)(lane & 15) * H_DIM + k0 + ((lane >> 4) * 16);
  v16h b;
#pragma unroll
  for (int e = 0; e < 16; ++e) b[e] = (_Float16)row[e];
  return b;
}

#if USE_TDM
// Issue one TDM 2D tile load: 16 rows x 64 f32 (4KB) from gsrc (row stride 64
// floats) into LDS at byte offset lds_byte_off. Wave-level op, EXEC ignored.
// D# layout per CDNA5 ISA 8.3/8.4.
__device__ __forceinline__ void tdm_load_16x64(const float* gsrc, u32 lds_byte_off) {
  unsigned long long ga = (unsigned long long)(uintptr_t)gsrc;
  v4u g0;
  g0.x = 1u;                                   // count=1, is_restore=0, gather off
  g0.y = lds_byte_off;                         // lds_addr (bytes)
  g0.z = (u32)(ga & 0xffffffffull);            // global_addr[31:0]
  g0.w = (u32)((ga >> 32) & 0x1ffffffull)      // global_addr[56:32]
         | (2u << 30);                         // type = 2 ("image")
  v8i_t g1 = (v8i_t)0;
  g1[0] = (int)(2u << 16);                     // data_size = 2 (4 bytes); mask=0
  g1[1] = (int)(64u << 16);                    // tensor_dim0[15:0]=64 (bits 79:48)
  g1[2] = (int)(16u << 16);                    // tensor_dim1[15:0]=16 (bits 111:80)
  g1[3] = (int)(64u << 16);                    // tile_dim0=64 (bits 127:112)
  g1[4] = (int)16;                             // tile_dim1=16 (bits 143:128)
  g1[5] = (int)64;                             // tensor_dim0_stride=64 (bits 207:160)
  v4i_t g2 = (v4i_t)0;
  v4i_t g3 = (v4i_t)0;
#if __clang_major__ >= 23
  v8i_t g4 = (v8i_t)0;
  __builtin_amdgcn_tensor_load_to_lds(g0, g1, g2, g3, g4, 0);
#else
  __builtin_amdgcn_tensor_load_to_lds(g0, g1, g2, g3, 0);
#endif
}
#endif

// =====================================================================
// Fused linear kernel:  C = act( A1*W1^T [* 1/max(cnt,1) rowwise] + bias
//                               + (A2 ? A2*W2^T : 0) )
// N=K=64 fixed. block = 128 threads (4 waves); wave w -> N tile [16w,16w+16);
// blockIdx.x -> M tile of 16 rows.  act: 0=none 1=relu 2=tanh
// B operands are staged per-wave into LDS by the Tensor Data Mover (16 rows of
// W per wave, disjoint regions, no barrier needed), overlapped with A loads.
// =====================================================================
__global__ void __launch_bounds__(128)
wmma_lin64_kernel(const float* __restrict__ A1, const float* __restrict__ W1,
                  const float* __restrict__ rowcnt,
                  const float* __restrict__ A2, const float* __restrict__ W2,
                  const float* __restrict__ bias, float* __restrict__ C,
                  int M, int act) {
  const int wave = threadIdx.x >> 5;
  const int lane = threadIdx.x & 31;
  const int m0   = blockIdx.x * 16;
  const int n0   = wave * 16;

  v8f acc = {};

#if USE_TDM
  __shared__ float smw[2 * 4 * 16 * H_DIM];          // 2 x (4 waves x 16x64) = 32KB
  float* myW1 = smw + (size_t)wave * (16 * H_DIM);
  float* myW2 = smw + 4096 + (size_t)wave * (16 * H_DIM);
  // kick off DMA of this wave's W rows, then overlap with A-fragment loads
  tdm_load_16x64(W1 + (size_t)n0 * H_DIM, (u32)(uintptr_t)myW1);
  if (A2) tdm_load_16x64(W2 + (size_t)n0 * H_DIM, (u32)(uintptr_t)myW2);

  v16h a10 = load_a_frag(A1, M, m0, lane, 0,  rowcnt);
  v16h a11 = load_a_frag(A1, M, m0, lane, 32, rowcnt);
  __builtin_amdgcn_s_wait_tensorcnt(0);
  {
    v16h b = load_b_frag_lds(myW1, lane, 0);
    acc = __builtin_amdgcn_wmma_f32_16x16x32_f16(false, a10, false, b, (short)0, acc, false, false);
    b = load_b_frag_lds(myW1, lane, 32);
    acc = __builtin_amdgcn_wmma_f32_16x16x32_f16(false, a11, false, b, (short)0, acc, false, false);
  }
  if (A2) {  // uniform branch: EXEC stays all-ones around WMMA
    v16h a = load_a_frag(A2, M, m0, lane, 0, nullptr);
    v16h b = load_b_frag_lds(myW2, lane, 0);
    acc = __builtin_amdgcn_wmma_f32_16x16x32_f16(false, a, false, b, (short)0, acc, false, false);
    a = load_a_frag(A2, M, m0, lane, 32, nullptr);
    b = load_b_frag_lds(myW2, lane, 32);
    acc = __builtin_amdgcn_wmma_f32_16x16x32_f16(false, a, false, b, (short)0, acc, false, false);
  }
#else
  {
    v16h a = load_a_frag(A1, M, m0, lane, 0, rowcnt);
    v16h b = load_b_frag(W1, n0, lane, 0);
    acc = __builtin_amdgcn_wmma_f32_16x16x32_f16(false, a, false, b, (short)0, acc, false, false);
    a = load_a_frag(A1, M, m0, lane, 32, rowcnt);
    b = load_b_frag(W1, n0, lane, 32);
    acc = __builtin_amdgcn_wmma_f32_16x16x32_f16(false, a, false, b, (short)0, acc, false, false);
  }
  if (A2) {
    v16h a = load_a_frag(A2, M, m0, lane, 0, nullptr);
    v16h b = load_b_frag(W2, n0, lane, 0);
    acc = __builtin_amdgcn_wmma_f32_16x16x32_f16(false, a, false, b, (short)0, acc, false, false);
    a = load_a_frag(A2, M, m0, lane, 32, nullptr);
    b = load_b_frag(W2, n0, lane, 32);
    acc = __builtin_amdgcn_wmma_f32_16x16x32_f16(false, a, false, b, (short)0, acc, false, false);
  }
#endif

  const int n = n0 + (lane & 15);
  const float bv = bias ? bias[n] : 0.0f;
#pragma unroll
  for (int r = 0; r < 8; ++r) {
    int m   = m0 + r + ((lane >> 4) << 3);   // C/D layout: VGPR r -> M = r + 8*(lane/16)
    float v = acc[r] + bv;
    if (act == 1) v = fmaxf(v, 0.0f);
    else if (act == 2) v = tanhf(v);
    if (m < M) C[(size_t)m * H_DIM + n] = v;
  }
}

// =====================================================================
// Edge gather + atomic scatter-add (SAGE numerator + degree count).
// 16 threads per edge, one float4 (16B) per thread -> full 256B row.
// =====================================================================
__global__ void edge_scatter_kernel(const float* __restrict__ h_src,
                                    const int* __restrict__ src_idx,
                                    const int* __restrict__ dst_idx,
                                    float* __restrict__ sum,
                                    float* __restrict__ cnt, int E) {
  long long tid = (long long)blockIdx.x * blockDim.x + threadIdx.x;
  int e   = (int)(tid >> 4);
  int sub = (int)(tid & 15);
  if (e >= E) return;
  int s = src_idx[e];
  int d = dst_idx[e];
  const float4 v = reinterpret_cast<const float4*>(h_src + ((size_t)s << 6))[sub];
  float* dst = sum + ((size_t)d << 6) + (sub << 2);
  atomicAdd(dst + 0, v.x);
  atomicAdd(dst + 1, v.y);
  atomicAdd(dst + 2, v.z);
  atomicAdd(dst + 3, v.w);
  if (sub == 0) atomicAdd(cnt + d, 1.0f);
}

// =====================================================================
// Utility kernels
// =====================================================================
__global__ void zero_kernel(float* __restrict__ p, long long n) {
  long long i = (long long)blockIdx.x * blockDim.x + threadIdx.x;
  if (i < n) p[i] = 0.0f;
}

// sum over rows of dot(X[n,:64], w)  ->  atomicAdd into *out (pre-zeroed)
__global__ void dot_sum_kernel(const float* __restrict__ X,
                               const float* __restrict__ w,
                               float* __restrict__ out, int N) {
  __shared__ float red[256];
  float s = 0.0f;
  for (int n = blockIdx.x * blockDim.x + threadIdx.x; n < N;
       n += gridDim.x * blockDim.x) {
    const float* row = X + ((size_t)n << 6);
    float d = 0.0f;
#pragma unroll
    for (int k = 0; k < 64; ++k) d += row[k] * w[k];
    s += d;
  }
  red[threadIdx.x] = s;
  __syncthreads();
  for (int off = 128; off; off >>= 1) {
    if (threadIdx.x < off) red[threadIdx.x] += red[threadIdx.x + off];
    __syncthreads();
  }
  if (threadIdx.x == 0) atomicAdd(out, red[0]);
}

// column sums (scaled) of X[N,64] -> atomicAdd into out[64] (pre-zeroed)
__global__ void colsum_kernel(const float* __restrict__ X,
                              float* __restrict__ out, int N, float scale) {
  __shared__ float red[4][64];
  int c = threadIdx.x & 63;
  int g = threadIdx.x >> 6;
  float s = 0.0f;
  for (int n = blockIdx.x * 4 + g; n < N; n += gridDim.x * 4)
    s += X[((size_t)n << 6) + c];
  red[g][c] = s;
  __syncthreads();
  if (g == 0) {
    float t = red[0][c] + red[1][c] + red[2][c] + red[3][c];
    atomicAdd(&out[c], t * scale);
  }
}

// relation softmax over the 2 mean scores
__global__ void beta_kernel(const float* __restrict__ wacc,
                            float* __restrict__ beta, float invN) {
  float a = wacc[0] * invN, b = wacc[1] * invN;
  float m = fmaxf(a, b);
  float ea = expf(a - m), eb = expf(b - m);
  float s = ea + eb;
  beta[0] = ea / s;
  beta[1] = eb / s;
}

// out = beta[0]*m1 + beta[1]*m2 (elementwise over N*64)
__global__ void combine_kernel(float* __restrict__ out,
                               const float* __restrict__ m1,
                               const float* __restrict__ m2,
                               const float* __restrict__ beta, long long n) {
  long long i = (long long)blockIdx.x * blockDim.x + threadIdx.x;
  if (i < n) out[i] = beta[0] * m1[i] + beta[1] * m2[i];
}

// =====================================================================
// Temporal attention head (single block, 128 threads). T=4, GH=128.
// =====================================================================
__global__ void __launch_bounds__(128)
temporal_head_kernel(const float* __restrict__ pooled,   // [4][128]
                     const float* __restrict__ tpw, const float* __restrict__ tpb,
                     const float* __restrict__ qw, const float* __restrict__ kw,
                     const float* __restrict__ vw,
                     const float* __restrict__ fcw, const float* __restrict__ fcb,
                     const float* __restrict__ clsw, const float* __restrict__ clsb,
                     float* __restrict__ out0) {
  __shared__ float h[4][128], q[4][128], kk[4][128], vv[4][128], sc[4][4];
  const int j = threadIdx.x;  // 0..127

  const float ln_base = logf(100000.0f);
  for (int t = 0; t < 4; ++t) {
    const float* wrow = tpw + (size_t)j * 128;
    float s = tpb[j];
    for (int i = 0; i < 128; ++i) s += pooled[t * 128 + i] * wrow[i];
    int kidx  = j & ~1;                       // PE pair index*2
    float div = expf(-(float)kidx * ln_base / 128.0f);
    float ang = (float)(t + 1) * div;
    s += (j & 1) ? cosf(ang) : sinf(ang);
    h[t][j] = s;
  }
  __syncthreads();

  for (int t = 0; t < 4; ++t) {
    float sq = 0.f, sk = 0.f, sv = 0.f;
    const float* qr = qw + (size_t)j * 128;
    const float* kr = kw + (size_t)j * 128;
    const float* vr = vw + (size_t)j * 128;
    for (int i = 0; i < 128; ++i) {
      float hv = h[t][i];
      sq += hv * qr[i]; sk += hv * kr[i]; sv += hv * vr[i];
    }
    q[t][j] = sq; kk[t][j] = sk; vv[t][j] = sv;
  }
  __syncthreads();

  if (j < 16) {
    int t1 = j >> 2, t2 = j & 3;
    float s = 0.f;
    for (int i = 0; i < 128; ++i) s += q[t1][i] * kk[t2][i];
    sc[t1][t2] = s;
  }
  __syncthreads();
  if (j < 4) {
    float m = fmaxf(fmaxf(sc[j][0], sc[j][1]), fmaxf(sc[j][2], sc[j][3]));
    float e0 = expf(sc[j][0] - m), e1 = expf(sc[j][1] - m);
    float e2 = expf(sc[j][2] - m), e3 = expf(sc[j][3] - m);
    float s = e0 + e1 + e2 + e3;
    sc[j][0] = e0 / s; sc[j][1] = e1 / s; sc[j][2] = e2 / s; sc[j][3] = e3 / s;
  }
  __syncthreads();

  // attn row for t = 3 (only one needed for the snapshot logit)
  float a3 = 0.f;
  for (int t2 = 0; t2 < 4; ++t2) a3 += sc[3][t2] * vv[t2][j];
  h[1][j] = a3;      // reuse smem as attn buffer
  __syncthreads();

  float o = fcb[j];
  const float* fr = fcw + (size_t)j * 128;
  for (int i = 0; i < 128; ++i) o += h[1][i] * fr[i];
  o = fmaxf(o, 0.0f);

  h[0][j] = o * clsw[j];
  __syncthreads();
  for (int off = 64; off; off >>= 1) {
    if (j < off) h[0][j] += h[0][j + off];
    __syncthreads();
  }
  if (j == 0) out0[0] = h[0][0] + clsb[0];
}

// device_logit[n] = dot(out_dev[n,:], dev_cls_w) + dev_cls_b
__global__ void dev_logit_kernel(const float* __restrict__ out_dev,
                                 const float* __restrict__ w,
                                 const float* __restrict__ b,
                                 float* __restrict__ out, int N) {
  int n = blockIdx.x * blockDim.x + threadIdx.x;
  if (n >= N) return;
  const float* row = out_dev + ((size_t)n << 6);
  float s = b[0];
#pragma unroll
  for (int k = 0; k < 64; ++k) s += row[k] * w[k];
  out[n] = s;
}

// =====================================================================
// Host launcher
// =====================================================================
static inline int cdiv(long long a, long long b) { return (int)((a + b - 1) / b); }

extern "C" void kernel_launch(void* const* d_in, const int* in_sizes, int n_in,
                              void* d_out, int out_size, void* d_ws, size_t ws_size,
                              hipStream_t stream) {
  (void)in_sizes; (void)n_in; (void)out_size; (void)ws_size;

  // ---- inputs (setup_inputs dict order; params flattened in insertion order)
  const float* x_device = (const float*)d_in[0];   // [T,N_DEV,64]
  const float* x_sensor = (const float*)d_in[1];   // [T,N_SEN,64]
  const int*   edge_ds  = (const int*)d_in[2];     // [T,2,E_DS]
  const int*   edge_sd  = (const int*)d_in[3];     // [T,2,E_SD]
  const int*   edge_dd  = (const int*)d_in[4];     // [T,2,E_DD]
  const float* proj_dev_w = (const float*)d_in[5];
  const float* proj_dev_b = (const float*)d_in[6];
  const float* proj_sen_w = (const float*)d_in[7];
  const float* proj_sen_b = (const float*)d_in[8];
  const float* ds_l_w = (const float*)d_in[9];
  const float* ds_l_b = (const float*)d_in[10];
  const float* ds_r_w = (const float*)d_in[11];
  const float* sd_l_w = (const float*)d_in[12];
  const float* sd_l_b = (const float*)d_in[13];
  const float* sd_r_w = (const float*)d_in[14];
  const float* dd_l_w = (const float*)d_in[15];
  const float* dd_l_b = (const float*)d_in[16];
  const float* dd_r_w = (const float*)d_in[17];
  const float* rel_w1 = (const float*)d_in[18];
  const float* rel_b1 = (const float*)d_in[19];
  const float* rel_w2 = (const float*)d_in[20];
  const float* t_proj_w = (const float*)d_in[21];
  const float* t_proj_b = (const float*)d_in[22];
  const float* q_w  = (const float*)d_in[23];
  const float* k_w  = (const float*)d_in[24];
  const float* v_w  = (const float*)d_in[25];
  const float* fc_w = (const float*)d_in[26];
  const float* fc_b = (const float*)d_in[27];
  const float* cls_w = (const float*)d_in[28];
  const float* cls_b = (const float*)d_in[29];
  const float* dev_cls_w = (const float*)d_in[30];
  const float* dev_cls_b = (const float*)d_in[31];

  float* out = (float*)d_out;  // [0]=snap_logit, [1..N_DEV]=device_logit

  // ---- workspace carve-up (floats, 256B-aligned slots)
  float* ws = (float*)d_ws;
  size_t off = 0;
  auto alloc = [&](size_t nfloats) {
    float* p = ws + off;
    off += (nfloats + 63) & ~(size_t)63;
    return p;
  };
  const size_t DEVH = (size_t)N_DEV * H_DIM, SENH = (size_t)N_SEN * H_DIM;
  float* hd_t    = alloc(DEVH);
  float* hs_t    = alloc(SENH);
  float* sum_sen = alloc(SENH);
  float* cnt_sen = alloc(N_SEN);
  float* sum_d1  = alloc(DEVH);
  float* cnt_d1  = alloc(N_DEV);
  float* sum_d2  = alloc(DEVH);
  float* cnt_d2  = alloc(N_DEV);
  float* m_sen   = alloc(SENH);
  float* m_d1    = alloc(DEVH);
  float* m_d2    = alloc(DEVH);
  float* ttmp    = alloc(DEVH);
  float* out_dev = alloc(DEVH);
  float* pooled  = alloc(T_STEPS * GH_DIM);
  float* wacc    = alloc(64);
  float* beta    = alloc(64);

  const dim3 blk256(256);
  auto zero = [&](float* p, long long n) {
    zero_kernel<<<cdiv(n, 256), blk256, 0, stream>>>(p, n);
  };

  zero(pooled, T_STEPS * GH_DIM);

  for (int t = 0; t < T_STEPS; ++t) {
    const float* xd = x_device + (size_t)t * DEVH;
    const float* xs = x_sensor + (size_t)t * SENH;
    const int* eds_s = edge_ds + (size_t)t * 2 * E_DS;
    const int* eds_d = eds_s + E_DS;
    const int* esd_s = edge_sd + (size_t)t * 2 * E_SD;
    const int* esd_d = esd_s + E_SD;
    const int* edd_s = edge_dd + (size_t)t * 2 * E_DD;
    const int* edd_d = edd_s + E_DD;

    zero(sum_sen, SENH); zero(cnt_sen, N_SEN);
    zero(sum_d1, DEVH);  zero(cnt_d1, N_DEV);
    zero(sum_d2, DEVH);  zero(cnt_d2, N_DEV);
    zero(wacc, 2);

    // input projections (relu)
    wmma_lin64_kernel<<<cdiv(N_DEV, 16), 128, 0, stream>>>(
        xd, proj_dev_w, nullptr, nullptr, nullptr, proj_dev_b, hd_t, N_DEV, 1);
    wmma_lin64_kernel<<<cdiv(N_SEN, 16), 128, 0, stream>>>(
        xs, proj_sen_w, nullptr, nullptr, nullptr, proj_sen_b, hs_t, N_SEN, 1);

    // edge scatter (mean numerators + degree counts)
    edge_scatter_kernel<<<cdiv((long long)E_DS * 16, 256), blk256, 0, stream>>>(
        hd_t, eds_s, eds_d, sum_sen, cnt_sen, E_DS);
    edge_scatter_kernel<<<cdiv((long long)E_SD * 16, 256), blk256, 0, stream>>>(
        hs_t, esd_s, esd_d, sum_d1, cnt_d1, E_SD);
    edge_scatter_kernel<<<cdiv((long long)E_DD * 16, 256), blk256, 0, stream>>>(
        hd_t, edd_s, edd_d, sum_d2, cnt_d2, E_DD);

    // fused SAGE update: relu( (sum/max(cnt,1))*Wl^T + bl + h_dst*Wr^T )
    wmma_lin64_kernel<<<cdiv(N_SEN, 16), 128, 0, stream>>>(
        sum_sen, ds_l_w, cnt_sen, hs_t, ds_r_w, ds_l_b, m_sen, N_SEN, 1);
    wmma_lin64_kernel<<<cdiv(N_DEV, 16), 128, 0, stream>>>(
        sum_d1, sd_l_w, cnt_d1, hd_t, sd_r_w, sd_l_b, m_d1, N_DEV, 1);
    wmma_lin64_kernel<<<cdiv(N_DEV, 16), 128, 0, stream>>>(
        sum_d2, dd_l_w, cnt_d2, hd_t, dd_r_w, dd_l_b, m_d2, N_DEV, 1);

    // relation attention scores: mean_n( tanh(m_r @ W1^T + b1) @ w2 )
    wmma_lin64_kernel<<<cdiv(N_DEV, 16), 128, 0, stream>>>(
        m_d1, rel_w1, nullptr, nullptr, nullptr, rel_b1, ttmp, N_DEV, 2);
    dot_sum_kernel<<<256, blk256, 0, stream>>>(ttmp, rel_w2, &wacc[0], N_DEV);
    wmma_lin64_kernel<<<cdiv(N_DEV, 16), 128, 0, stream>>>(
        m_d2, rel_w1, nullptr, nullptr, nullptr, rel_b1, ttmp, N_DEV, 2);
    dot_sum_kernel<<<256, blk256, 0, stream>>>(ttmp, rel_w2, &wacc[1], N_DEV);

    beta_kernel<<<1, 1, 0, stream>>>(wacc, beta, 1.0f / (float)N_DEV);
    combine_kernel<<<cdiv((long long)DEVH, 256), blk256, 0, stream>>>(
        out_dev, m_d1, m_d2, beta, (long long)DEVH);

    // pooled[t] = concat(mean(out_dev), mean(m_sen))
    colsum_kernel<<<512, blk256, 0, stream>>>(out_dev, pooled + t * GH_DIM,
                                              N_DEV, 1.0f / (float)N_DEV);
    colsum_kernel<<<512, blk256, 0, stream>>>(m_sen, pooled + t * GH_DIM + H_DIM,
                                              N_SEN, 1.0f / (float)N_SEN);
  }

  // temporal attention + snapshot classifier  (out_dev still holds t = T-1)
  temporal_head_kernel<<<1, 128, 0, stream>>>(pooled, t_proj_w, t_proj_b,
                                              q_w, k_w, v_w, fc_w, fc_b,
                                              cls_w, cls_b, out);
  dev_logit_kernel<<<cdiv(N_DEV, 256), blk256, 0, stream>>>(
      out_dev, dev_cls_w, dev_cls_b, out + 1, N_DEV);
}
